// RandomWalkGAT_3848290697850
// MI455X (gfx1250) — matrix operand order repaired and checked by reading, not compiled, not run on hardware
//
#include <hip/hip_runtime.h>
#include <hip/hip_bf16.h>

// ---------------- problem constants (match reference) ----------------
#define NN        50000      // nodes
#define NE        800000     // edges
#define F_IN      128
#define HID       32
#define HEADS     8
#define H1        256        // HEADS*HID
#define NUM_WALKS 512
#define WALK_LEN  64
#define WINDOW    5
#define NEG_S     10
#define TEMP      0.07f
#define NEG_SLOPE 0.2f

#define CDIV(a, b) (((a) + (b) - 1) / (b))

typedef __attribute__((ext_vector_type(16))) __bf16 v16bf;
typedef __attribute__((ext_vector_type(8)))  __bf16 v8bf;   // 16 bytes -> b128 load
typedef __attribute__((ext_vector_type(8)))  float  v8f;

// ---------------- small utility kernels ----------------
__global__ void fill_f32_kernel(float* p, float v, long n) {
  long i = (long)blockIdx.x * blockDim.x + threadIdx.x;
  if (i < n) p[i] = v;
}
__global__ void fill_u32_kernel(unsigned* p, unsigned v, long n) {
  long i = (long)blockIdx.x * blockDim.x + threadIdx.x;
  if (i < n) p[i] = v;
}
__global__ void f32_to_bf16_kernel(const float* __restrict__ in,
                                   __bf16* __restrict__ out, long n) {
  long i = (long)blockIdx.x * blockDim.x + threadIdx.x;
  if (i < n) out[i] = (__bf16)in[i];
}

// Pack weight W[K,N] (f32 row-major) into per-lane WMMA B-fragment order:
// Bp[((tn*(K/32) + ks)*32 + lane)*16 + e] = bf16(W[(ks*32 + e + 8*half + (e>=8 ? 8 : 0))*N + tn*16 + mr])
// so the GEMM reads one contiguous, 32B-aligned v16bf per lane per K-step.
__global__ void pack_b_wmma_kernel(const float* __restrict__ W,
                                   __bf16* __restrict__ Bp, int K, int N) {
  int idx = blockIdx.x * blockDim.x + threadIdx.x;
  int total = (N / 16) * (K / 32) * 32 * 16;
  if (idx >= total) return;
  int e    = idx & 15;
  int lane = (idx >> 4) & 31;
  int rest = idx >> 9;            // tn*(K/32) + ks
  int ks   = rest % (K / 32);
  int tn   = rest / (K / 32);
  int half = lane >> 4, mr = lane & 15;
  int k = ks * 32 + e + 8 * half + ((e >= 8) ? 8 : 0);
  int n = tn * 16 + mr;
  Bp[idx] = (__bf16)W[(size_t)k * N + n];
}

// monotonic float <-> uint encoding for atomic max on floats
__device__ __forceinline__ unsigned f2o(float f) {
  unsigned u = __float_as_uint(f);
  return (u & 0x80000000u) ? ~u : (u | 0x80000000u);
}
__device__ __forceinline__ float o2f(unsigned e) {
  return __uint_as_float((e & 0x80000000u) ? (e & 0x7fffffffu) : ~e);
}
#define ENC_NEG_INF 0x007FFFFFu   // f2o(-INF)

// ---------------- WMMA bf16 GEMM: C[M,N] = A[M,K] * B[K,N] ----------------
// One wave computes a 16x64 strip: 4 accumulators share one A fragment per
// K-step (2 A-loads + 8 B-loads -> 4 WMMAs). K templated -> fully unrolled
// chain of K/32*4 v_wmma ops. EXEC all-ones (exact grid, no divergence).
template <int K>
__global__ __launch_bounds__(32)
void gemm_bf16_wmma_kernel(const __bf16* __restrict__ A,
                           const __bf16* __restrict__ Bp,
                           float* __restrict__ C, int M, int N) {
  const int nblk = N >> 6;             // 64-wide column blocks
  const int tm   = blockIdx.x / nblk;
  const int tb   = blockIdx.x % nblk;
  const int lane = threadIdx.x;
  const int half = lane >> 4;
  const int mr   = lane & 15;

  const __bf16* Arow = A + (size_t)(tm * 16 + mr) * K;
  const size_t tileStride = (size_t)(K / 32) * 32 * 16;  // elems per 16-wide N-tile
  const __bf16* Bb[4];
#pragma unroll
  for (int j = 0; j < 4; ++j)
    Bb[j] = Bp + (size_t)(tb * 4 + j) * tileStride + (size_t)lane * 16;

  v8f acc[4] = {{}, {}, {}, {}};
#pragma unroll
  for (int ks = 0; ks < K / 32; ++ks) {
    const int k0 = ks * 32;
    // A fragment: e<8 -> K=k0+e+8*half ; e>=8 -> K=k0+16+(e-8)+8*half
    v8bf alo = *(const v8bf*)(Arow + k0 + 8 * half);
    v8bf ahi = *(const v8bf*)(Arow + k0 + 16 + 8 * half);
    v16bf a = __builtin_shufflevector(alo, ahi, 0, 1, 2, 3, 4, 5, 6, 7,
                                      8, 9, 10, 11, 12, 13, 14, 15);
#pragma unroll
    for (int j = 0; j < 4; ++j) {
      v16bf b = *(const v16bf*)(Bb[j] + (size_t)ks * 32 * 16);
      acc[j] = __builtin_amdgcn_wmma_f32_16x16x32_bf16(
          /*neg_a=*/false, a, /*neg_b=*/false, b,
          /*c_mod=*/(short)0, acc[j], /*reuse_a=*/false, /*reuse_b=*/false);
    }
  }
  // C/D layout: VGPR r -> M = r + 8*half, N = lane&15 within each 16-wide tile
#pragma unroll
  for (int j = 0; j < 4; ++j)
#pragma unroll
    for (int r = 0; r < 8; ++r)
      C[(size_t)(tm * 16 + r + 8 * half) * N + (tb * 4 + j) * 16 + mr] = acc[j][r];
}

// ---------------- per-node attention logits ----------------
__global__ void node_al_kernel(const float* __restrict__ xw,
                               const float* __restrict__ a_src,
                               const float* __restrict__ a_dst,
                               float* __restrict__ als, float* __restrict__ ald,
                               int n_nodes, int heads, int ch) {
  int idx = blockIdx.x * blockDim.x + threadIdx.x;
  if (idx >= n_nodes * heads) return;
  int n = idx / heads, h = idx % heads;
  const float* xr = xw + ((size_t)n * heads + h) * ch;
  const float* as = a_src + (size_t)h * ch;
  const float* ad = a_dst + (size_t)h * ch;
  float ss = 0.f, sd = 0.f;
  for (int c = 0; c < ch; ++c) { float v = xr[c]; ss += v * as[c]; sd += v * ad[c]; }
  als[idx] = ss;
  ald[idx] = sd;
}

__device__ __forceinline__ void edge_endpoints(const int* ei, int e, int n_edges,
                                               int& s, int& d) {
  if (e < n_edges) { s = ei[e]; d = ei[n_edges + e]; }
  else             { s = d = e - n_edges; }   // self-loop
}

// ---------------- edge pass 1: segment max of leaky_relu(als[src]+ald[dst]) ----
__global__ void edge_max_kernel(const int* __restrict__ ei,
                                const float* __restrict__ als,
                                const float* __restrict__ ald,
                                unsigned* __restrict__ menc,
                                int n_edges, int n_nodes, int heads) {
  long idx = (long)blockIdx.x * blockDim.x + threadIdx.x;
  long tot = (long)(n_edges + n_nodes) * heads;
  if (idx >= tot) return;
  int e = (int)(idx / heads), h = (int)(idx % heads);
  int s, d; edge_endpoints(ei, e, n_edges, s, d);
  float v = als[(size_t)s * heads + h] + ald[(size_t)d * heads + h];
  v = (v > 0.f) ? v : NEG_SLOPE * v;
  atomicMax(&menc[(size_t)d * heads + h], f2o(v));
}

// ---------------- edge pass 2 (channel-parallel): -----------------------------
// thread (e,h,c): ex = exp(lrelu - m[d,h]); acc[d,h,c] += ex*xw[s,h,c];
// consecutive lanes cover consecutive channels -> coalesced gather + scatter.
__global__ void edge_accum_kernel(const int* __restrict__ ei,
                                  const float* __restrict__ als,
                                  const float* __restrict__ ald,
                                  const unsigned* __restrict__ menc,
                                  const float* __restrict__ xw,
                                  float* __restrict__ den,
                                  float* __restrict__ acc,
                                  int n_edges, int n_nodes, int heads, int ch) {
  long idx = (long)blockIdx.x * blockDim.x + threadIdx.x;
  long tot = (long)(n_edges + n_nodes) * heads * ch;
  if (idx >= tot) return;
  int  c  = (int)(idx % ch);
  long eh = idx / ch;
  int  h  = (int)(eh % heads);
  int  e  = (int)(eh / heads);
  int s, d; edge_endpoints(ei, e, n_edges, s, d);
  float v = als[(size_t)s * heads + h] + ald[(size_t)d * heads + h];
  v = (v > 0.f) ? v : NEG_SLOPE * v;
  float ex = __expf(v - o2f(menc[(size_t)d * heads + h]));
  if (c == 0) atomicAdd(&den[(size_t)d * heads + h], ex);
  atomicAdd(&acc[((size_t)d * heads + h) * ch + c],
            ex * xw[((size_t)s * heads + h) * ch + c]);
}

// ---------------- node finalize: out = acc/den + bias (+ optional ELU) --------
__global__ void finalize_kernel(const float* __restrict__ acc,
                                const float* __restrict__ den,
                                const float* __restrict__ bias,
                                float* __restrict__ out,
                                long total, int heads, int ch, int do_elu) {
  long idx = (long)blockIdx.x * blockDim.x + threadIdx.x;
  if (idx >= total) return;
  int hc = (int)(idx % (heads * ch));
  float v = acc[idx] / den[idx / ch] + bias[hc];
  if (do_elu) v = (v > 0.f) ? v : (__expf(v) - 1.f);
  out[idx] = v;
}

// ---------------- row norms of concat embedding (clipped at 1e-8) -------------
__global__ void node_norm_kernel(const float* __restrict__ emb1,
                                 const float* __restrict__ emb2,
                                 float* __restrict__ norms, int n_nodes) {
  int i = blockIdx.x * blockDim.x + threadIdx.x;
  if (i >= n_nodes) return;
  const float* p1 = emb1 + (size_t)i * H1;
  const float* p2 = emb2 + (size_t)i * H1;
  float s = 0.f;
  for (int c = 0; c < H1; ++c) { float v = p1[c]; s += v * v; }
  for (int c = 0; c < H1; ++c) { float v = p2[c]; s += v * v; }
  float nm = sqrtf(s);
  norms[i] = (nm > 1e-8f) ? nm : 1e-8f;
}

// ---------------- walk contrastive loss: one wave32 per anchor ----------------
__device__ __forceinline__ float wave_dot_concat(const float av[16],
                                                 const float* __restrict__ emb1,
                                                 const float* __restrict__ emb2,
                                                 int node, int lane) {
  const float* c1 = emb1 + (size_t)node * H1;
  const float* c2 = emb2 + (size_t)node * H1;
  float d = 0.f;
#pragma unroll
  for (int i = 0; i < 8; ++i) d += av[i]     * c1[lane + 32 * i];
#pragma unroll
  for (int i = 0; i < 8; ++i) d += av[8 + i] * c2[lane + 32 * i];
#pragma unroll
  for (int o = 16; o > 0; o >>= 1) d += __shfl_xor(d, o, 32);  // all lanes get sum
  return d;
}

__global__ __launch_bounds__(32)
void walk_loss_kernel(const float* __restrict__ emb1, const float* __restrict__ emb2,
                      const float* __restrict__ norms,
                      const int* __restrict__ walks, const int* __restrict__ negs,
                      float* __restrict__ out) {
  int anchor = blockIdx.x;                  // 0 .. NUM_WALKS*WALK_LEN-1
  int b = anchor / WALK_LEN, l = anchor % WALK_LEN;
  int lane = threadIdx.x;
  int a = walks[anchor];
  float sa = 1.f / norms[a];
  float av[16];
  const float* e1a = emb1 + (size_t)a * H1;
  const float* e2a = emb2 + (size_t)a * H1;
#pragma unroll
  for (int i = 0; i < 8; ++i) {
    av[i]     = e1a[lane + 32 * i] * sa;
    av[8 + i] = e2a[lane + 32 * i] * sa;
  }
  int  pnode[2 * WINDOW];
  bool pvalid[2 * WINDOW];
  bool has_pos = false;
#pragma unroll
  for (int p = 0; p < 2 * WINDOW; ++p) {
    int offv = (p < WINDOW) ? (p - WINDOW) : (p - WINDOW + 1);  // -5..-1, 1..5
    int pp = l + offv;
    bool v = (pp >= 0) && (pp < WALK_LEN);
    int ppc = pp < 0 ? 0 : (pp > WALK_LEN - 1 ? WALK_LEN - 1 : pp);
    pnode[p]  = walks[b * WALK_LEN + ppc];
    pvalid[p] = v;
    has_pos |= v;
  }
  float pos_sum = 0.f;
  for (int p = 0; p < 2 * WINDOW; ++p) {
    float d = wave_dot_concat(av, emb1, emb2, pnode[p], lane);
    float sim = d / (norms[pnode[p]] * TEMP);
    pos_sum += pvalid[p] ? __expf(sim) : 0.f;
  }
  float neg_sum = 0.f;
  for (int s = 0; s < NEG_S; ++s) {
    int nn = negs[anchor * NEG_S + s];
    bool inpos = false;
#pragma unroll
    for (int p = 0; p < 2 * WINDOW; ++p) inpos |= pvalid[p] && (nn == pnode[p]);
    float d = wave_dot_concat(av, emb1, emb2, nn, lane);
    float sim = d / (norms[nn] * TEMP);
    neg_sum += inpos ? 0.f : __expf(sim);
  }
  float safe = has_pos ? pos_sum : 1.f;
  float term = -__logf(safe / (safe + neg_sum));
  if (lane == 0) atomicAdd(out, has_pos ? term : 0.f);
}

// ---------------- launcher ----------------
extern "C" void kernel_launch(void* const* d_in, const int* in_sizes, int n_in,
                              void* d_out, int out_size, void* d_ws, size_t ws_size,
                              hipStream_t stream) {
  const float* x      = (const float*)d_in[0];
  const int*   ei     = (const int*)  d_in[1];
  const int*   walks  = (const int*)  d_in[2];
  const int*   negs   = (const int*)  d_in[3];
  const float* W1     = (const float*)d_in[4];
  const float* a_src1 = (const float*)d_in[5];
  const float* a_dst1 = (const float*)d_in[6];
  const float* b1     = (const float*)d_in[7];
  const float* W2     = (const float*)d_in[8];
  const float* a_src2 = (const float*)d_in[9];
  const float* a_dst2 = (const float*)d_in[10];
  const float* b2     = (const float*)d_in[11];
  float* out = (float*)d_out;

  char* ws = (char*)d_ws;
  size_t off = 0;
  auto alloc = [&](size_t bytes) {
    void* p = ws + off;
    off += (bytes + 255) & ~(size_t)255;
    return p;
  };
  __bf16*   xbf   = (__bf16*)  alloc((size_t)NN * F_IN * 2);
  __bf16*   w1bp  = (__bf16*)  alloc((size_t)F_IN * H1 * 2);   // packed B layout
  __bf16*   w2bp  = (__bf16*)  alloc((size_t)H1 * H1 * 2);     // packed B layout
  float*    xw    = (float*)   alloc((size_t)NN * H1 * 4);     // shared by both layers
  float*    als   = (float*)   alloc((size_t)NN * HEADS * 4);
  float*    ald   = (float*)   alloc((size_t)NN * HEADS * 4);
  unsigned* menc  = (unsigned*)alloc((size_t)NN * HEADS * 4);
  float*    den   = (float*)   alloc((size_t)NN * HEADS * 4);
  float*    acc   = (float*)   alloc((size_t)NN * H1 * 4);     // shared by both layers
  float*    emb1  = (float*)   alloc((size_t)NN * H1 * 4);
  float*    emb2  = (float*)   alloc((size_t)NN * H1 * 4);
  float*    norms = (float*)   alloc((size_t)NN * 4);
  // emb1 in bf16 only lives between layer-1 finalize and layer-2 GEMM: alias acc.
  __bf16* emb1bf = (__bf16*)acc;

  const int ET = NE + NN;  // edges incl. self-loops
  const int TPB = 256;

  // ---- convert / pack operands for the WMMA GEMMs ----
  {
    long n = (long)NN * F_IN;
    f32_to_bf16_kernel<<<CDIV(n, TPB), TPB, 0, stream>>>(x, xbf, n);
    pack_b_wmma_kernel<<<CDIV(F_IN * H1, TPB), TPB, 0, stream>>>(W1, w1bp, F_IN, H1);
    pack_b_wmma_kernel<<<CDIV(H1 * H1, TPB), TPB, 0, stream>>>(W2, w2bp, H1, H1);
  }

  // ================= layer 1 (8 heads x 32 ch) =================
  gemm_bf16_wmma_kernel<F_IN><<<(NN / 16) * (H1 / 64), 32, 0, stream>>>(
      xbf, w1bp, xw, NN, H1);

  node_al_kernel<<<CDIV(NN * HEADS, TPB), TPB, 0, stream>>>(
      xw, a_src1, a_dst1, als, ald, NN, HEADS, HID);

  fill_u32_kernel<<<CDIV((long)NN * HEADS, TPB), TPB, 0, stream>>>(
      menc, ENC_NEG_INF, (long)NN * HEADS);
  fill_f32_kernel<<<CDIV((long)NN * HEADS, TPB), TPB, 0, stream>>>(
      den, 0.f, (long)NN * HEADS);
  fill_f32_kernel<<<CDIV((long)NN * H1, TPB), TPB, 0, stream>>>(
      acc, 0.f, (long)NN * H1);

  edge_max_kernel<<<CDIV((long)ET * HEADS, TPB), TPB, 0, stream>>>(
      ei, als, ald, menc, NE, NN, HEADS);
  edge_accum_kernel<<<CDIV((long)ET * HEADS * HID, TPB), TPB, 0, stream>>>(
      ei, als, ald, menc, xw, den, acc, NE, NN, HEADS, HID);

  finalize_kernel<<<CDIV((long)NN * H1, TPB), TPB, 0, stream>>>(
      acc, den, b1, emb1, (long)NN * H1, HEADS, HID, /*elu=*/1);

  // ================= layer 2 (1 head x 256 ch) =================
  {
    long n = (long)NN * H1;
    f32_to_bf16_kernel<<<CDIV(n, TPB), TPB, 0, stream>>>(emb1, emb1bf, n);
  }
  gemm_bf16_wmma_kernel<H1><<<(NN / 16) * (H1 / 64), 32, 0, stream>>>(
      emb1bf, w2bp, xw, NN, H1);

  node_al_kernel<<<CDIV(NN, TPB), TPB, 0, stream>>>(
      xw, a_src2, a_dst2, als, ald, NN, 1, H1);

  fill_u32_kernel<<<CDIV((long)NN, TPB), TPB, 0, stream>>>(
      menc, ENC_NEG_INF, (long)NN);
  fill_f32_kernel<<<CDIV((long)NN, TPB), TPB, 0, stream>>>(den, 0.f, (long)NN);
  fill_f32_kernel<<<CDIV((long)NN * H1, TPB), TPB, 0, stream>>>(
      acc, 0.f, (long)NN * H1);   // overwrites emb1bf (done with it)

  edge_max_kernel<<<CDIV((long)ET, TPB), TPB, 0, stream>>>(
      ei, als, ald, menc, NE, NN, 1);
  edge_accum_kernel<<<CDIV((long)ET * H1, TPB), TPB, 0, stream>>>(
      ei, als, ald, menc, xw, den, acc, NE, NN, 1, H1);

  finalize_kernel<<<CDIV((long)NN * H1, TPB), TPB, 0, stream>>>(
      acc, den, b2, emb2, (long)NN * H1, 1, H1, /*elu=*/0);

  // ================= walk contrastive loss =================
  node_norm_kernel<<<CDIV(NN, TPB), TPB, 0, stream>>>(emb1, emb2, norms, NN);
  fill_f32_kernel<<<1, 32, 0, stream>>>(out, 0.f, out_size);
  walk_loss_kernel<<<NUM_WALKS * WALK_LEN, 32, 0, stream>>>(
      emb1, emb2, norms, walks, negs, out);
}
